// NN_log_den_gauss_27281632264836
// MI455X (gfx1250) — matrix operand order repaired and checked by reading, not compiled, not run on hardware
//
#include <hip/hip_runtime.h>

// out[i] = -( (X[i]-mean) @ prec @ (X[i]-mean)^T ), n=65536 rows, d=256.
//
// MI455X plan: GEMM Z = Y@P on the bf16 matrix cores (v_wmma_f32_16x16x32_bf16)
// with fp32 split into bf16 hi+lo and 3 products (AhBh + AhBl + AlBh) for
// near-fp32 accuracy at bf16-core throughput. HBM floor ~2.9us (64MB X @
// 23.3TB/s); prec (256KB) stays resident in the 192MB L2.
//
// Data movement:
//  - X staged with global_load_b128 (float4/thread) + ds_store_b64 (4x bf16).
//  - A (16x32 bf16) fragment per lane = two contiguous 16B runs of a
//    row-major Y row -> 2x ds_load_b128, hoisted to registers for all K.
//  - B (32x16 bf16) fragment per lane = two contiguous 16B runs of a prec
//    column -> prec tile staged TRANSPOSED [16 cols][256 K] -> 2x ds_load_b128.
//  - Row strides padded to 264 elems (528B = 132 dwords) so b128 ops hit
//    every LDS bank exactly twice (the floor for 512B/instruction).

typedef __attribute__((ext_vector_type(16))) __bf16        v16bf;
typedef __attribute__((ext_vector_type(4)))  __bf16        v4bf;
typedef __attribute__((ext_vector_type(8)))  float         v8f;
typedef __attribute__((ext_vector_type(4)))  float         v4f;
typedef __attribute__((ext_vector_type(4)))  unsigned int  u32x4;

union Frag16 {
    v16bf  v;
    __bf16 e[16];
    u32x4  q[2];   // two 16B halves -> ds_load_b128 each
};

#define ROWS_PER_BLOCK 128
#define DDIM   256
#define NTILE  16    // 256 / 16 column tiles
#define KSTEPS 8     // 256 / 32 K steps
#define LDSY   (DDIM + 8)   // padded row stride, elements (528B)
#define LDSP   (DDIM + 8)

__global__ __launch_bounds__(256)
void quad_form_bf16wmma_kernel(const float* __restrict__ X,
                               const float* __restrict__ mean,
                               const float* __restrict__ prec,
                               float* __restrict__ out,
                               int n_rows) {
    extern __shared__ char smem_raw[];
    __bf16* lds_yh  = (__bf16*)smem_raw;                 // [128][LDSY]
    __bf16* lds_yl  = lds_yh + ROWS_PER_BLOCK * LDSY;    // [128][LDSY]
    __bf16* lds_pth = lds_yl + ROWS_PER_BLOCK * LDSY;    // [16][LDSP] transposed
    __bf16* lds_ptl = lds_pth + 16 * LDSP;               // [16][LDSP]

    const int t    = threadIdx.x;      // 0..255
    const int lane = t & 31;
    const int w    = t >> 5;           // wave id 0..7
    const int half = lane >> 4;        // 0: lanes 0-15, 1: lanes 16-31
    const int nl   = lane & 15;
    const long long row0 = (long long)blockIdx.x * ROWS_PER_BLOCK;

    // ---- Stage Y = X - mean into LDS as bf16 hi/lo ----
    // 256 threads cover 4 rows x 64 float4 per iteration:
    //   global_load_b128 (coalesced) -> subtract/split -> ds_store_b64.
    {
        const int sr = t >> 6;         // sub-row 0..3
        const int sc = (t & 63) * 4;   // column base 0..252
        const v4f mv4 = *(const v4f*)(mean + sc);
        for (int i = 0; i < ROWS_PER_BLOCK / 4; ++i) {
            int rloc = i * 4 + sr;
            long long r = row0 + rloc;
            v4f x = {};
            if (r < n_rows) x = *(const v4f*)(X + r * DDIM + sc);
            v4bf yh4, yl4;
#pragma unroll
            for (int e = 0; e < 4; ++e) {
                float y = x[e] - mv4[e];
                __bf16 yh = (__bf16)y;
                yh4[e] = yh;
                yl4[e] = (__bf16)(y - (float)yh);
            }
            *(v4bf*)(lds_yh + rloc * LDSY + sc) = yh4;
            *(v4bf*)(lds_yl + rloc * LDSY + sc) = yl4;
        }
    }
    __syncthreads();

    // ---- Hoist A-fragments (this wave's 16 rows, all K) into registers ----
    // Per lane: e[0..7]  = K in [32k + 8*half,      +8)  -> one b128
    //           e[8..15] = K in [32k + 16 + 8*half, +8)  -> one b128
    Frag16 Ah[KSTEPS], Al[KSTEPS];
    const int arow = 16 * w + nl;
    const __bf16* yrh = lds_yh + arow * LDSY;
    const __bf16* yrl = lds_yl + arow * LDSY;
#pragma unroll
    for (int k = 0; k < KSTEPS; ++k) {
        int off = 32 * k + 8 * half;
        Ah[k].q[0] = *(const u32x4*)(yrh + off);
        Ah[k].q[1] = *(const u32x4*)(yrh + off + 16);
        Al[k].q[0] = *(const u32x4*)(yrl + off);
        Al[k].q[1] = *(const u32x4*)(yrl + off + 16);
    }

    float fsum[8];
#pragma unroll
    for (int v = 0; v < 8; ++v) fsum[v] = 0.0f;

    // ---- Loop over 16-column tiles of prec ----
    for (int j = 0; j < NTILE; ++j) {
        __syncthreads();  // previous tile's consumers done
        // Stage prec[:, 16j:16j+16] TRANSPOSED into LDS: pt[cc][kk].
        // Thread t reads the 16-col segment of prec row t (64B contiguous).
        {
            const float* psrc = prec + (size_t)t * DDIM + j * 16;
#pragma unroll
            for (int c = 0; c < 16; ++c) {
                float p = psrc[c];
                __bf16 ph = (__bf16)p;
                __bf16 pl = (__bf16)(p - (float)ph);
                lds_pth[c * LDSP + t] = ph;
                lds_ptl[c * LDSP + t] = pl;
            }
        }
        __syncthreads();

        v8f acc = {};
        const __bf16* prh = lds_pth + nl * LDSP;
        const __bf16* prl = lds_ptl + nl * LDSP;
#pragma unroll
        for (int k = 0; k < KSTEPS; ++k) {
            // B per lane: e[0..7] = K in [32k + 8*half, +8), e[8..15] = K in [32k + 16 + 8*half, +8), col = nl
            Frag16 bh, bl;
            int off = 32 * k + 8 * half;
            bh.q[0] = *(const u32x4*)(prh + off);
            bh.q[1] = *(const u32x4*)(prh + off + 16);
            bl.q[0] = *(const u32x4*)(prl + off);
            bl.q[1] = *(const u32x4*)(prl + off + 16);
            // acc += Ah*Bh + Ah*Bl + Al*Bh  (fp32 accumulate)
            acc = __builtin_amdgcn_wmma_f32_16x16x32_bf16(
                false, Ah[k].v, false, bh.v, (short)0, acc, false, false);
            acc = __builtin_amdgcn_wmma_f32_16x16x32_bf16(
                false, Ah[k].v, false, bl.v, (short)0, acc, false, false);
            acc = __builtin_amdgcn_wmma_f32_16x16x32_bf16(
                false, Al[k].v, false, bh.v, (short)0, acc, false, false);
        }

        // ---- fsum[m] += Z[m][n] * Y[m][n] for this column tile ----
        // f32 C 16x16 layout: lane holds (m = v + 8*half, n = lane%16)
#pragma unroll
        for (int v = 0; v < 8; ++v) {
            int rl  = 16 * w + 8 * half + v;
            int col = j * 16 + nl;
            float y = (float)lds_yh[rl * LDSY + col] +
                      (float)lds_yl[rl * LDSY + col];
            fsum[v] += acc[v] * y;
        }
    }

    // ---- Reduce across the 16 lanes of each half, write -sum ----
#pragma unroll
    for (int v = 0; v < 8; ++v) {
        float s = fsum[v];
        s += __shfl_xor(s, 1, 32);
        s += __shfl_xor(s, 2, 32);
        s += __shfl_xor(s, 4, 32);
        s += __shfl_xor(s, 8, 32);
        if (nl == 0) {
            long long r = row0 + 16 * w + 8 * half + v;
            if (r < n_rows) out[r] = -s;
        }
    }
}

extern "C" void kernel_launch(void* const* d_in, const int* in_sizes, int n_in,
                              void* d_out, int out_size, void* d_ws, size_t ws_size,
                              hipStream_t stream) {
    const float* X    = (const float*)d_in[0];  // [n, 256]
    const float* mean = (const float*)d_in[1];  // [1, 256]
    const float* prec = (const float*)d_in[2];  // [256, 256]
    float* out = (float*)d_out;                 // [n]

    const int n_rows = in_sizes[0] / DDIM;      // 65536
    const int blocks = (n_rows + ROWS_PER_BLOCK - 1) / ROWS_PER_BLOCK;
    const size_t shmem =
        (size_t)(2 * ROWS_PER_BLOCK * LDSY + 2 * 16 * LDSP) * sizeof(__bf16); // ~148.5 KB

    hipLaunchKernelGGL(quad_form_bf16wmma_kernel, dim3(blocks), dim3(256),
                       shmem, stream, X, mean, prec, out, n_rows);
}
